// MultiHeadAttention_62440234549886
// MI455X (gfx1250) — matrix operand order repaired and checked by reading, not compiled, not run on hardware
//
#include <hip/hip_runtime.h>
#include <hip/hip_bf16.h>

#define D_MODEL 1024
#define N_HEADS 16
#define D_HEAD  64
#define BATCH   4
#define SEQ     2048

typedef __attribute__((ext_vector_type(16))) _Float16 v16h;
typedef __attribute__((ext_vector_type(8)))  _Float16 v8h;
typedef __attribute__((ext_vector_type(2)))  _Float16 v2h;
typedef __attribute__((ext_vector_type(8)))  float    v8f;
typedef __attribute__((ext_vector_type(4)))  float    v4f;

__device__ __forceinline__ v8f wmma_f16(v16h a, v16h b, v8f c) {
  // D = A(16x32 f16) * B(32x16 f16) + C(16x16 f32)
  return __builtin_amdgcn_wmma_f32_16x16x32_f16(
      /*neg_a=*/false, a, /*neg_b=*/false, b,
      /*c_mod=*/(short)0, c, /*reuse_a=*/false, /*reuse_b=*/false);
}

// Packed f32->f16 conversion (v_cvt_pk_f16_f32): 2 elements per VALU op.
__device__ __forceinline__ void put4(v16h& r, int base, v4f x) {
  v2h lo = __builtin_bit_cast(v2h, __builtin_amdgcn_cvt_pkrtz(x[0], x[1]));
  v2h hi = __builtin_bit_cast(v2h, __builtin_amdgcn_cvt_pkrtz(x[2], x[3]));
  r[base + 0] = lo[0]; r[base + 1] = lo[1];
  r[base + 2] = hi[0]; r[base + 3] = hi[1];
}

// Build a 16-half fragment from two contiguous 8-element runs (f32 source).
__device__ __forceinline__ v16h frag16(const float* p0, const float* p1) {
  v4f a0 = *(const v4f*)(p0);
  v4f a1 = *(const v4f*)(p0 + 4);
  v4f b0 = *(const v4f*)(p1);
  v4f b1 = *(const v4f*)(p1 + 4);
  v16h r;
  put4(r, 0, a0); put4(r, 4, a1); put4(r, 8, b0); put4(r, 12, b1);
  return r;
}

// Same, from f16 memory (16B vector loads).
__device__ __forceinline__ v16h frag16(const _Float16* p0, const _Float16* p1) {
  v8h a = *(const v8h*)p0;
  v8h b = *(const v8h*)p1;
  v16h r;
#pragma unroll
  for (int i = 0; i < 8; ++i) { r[i] = a[i]; r[8 + i] = b[i]; }
  return r;
}

// 16-lane-row max reduction. v_permlane16_b32 is pure VALU (no LDS round
// trip like ds_bpermute) and operates independently on each 16-lane half of
// the wave32 — exactly the C-fragment row striping.
#if __has_builtin(__builtin_amdgcn_permlane16)
__device__ __forceinline__ float perm16(float v, unsigned s0, unsigned s1) {
  unsigned u = __builtin_bit_cast(unsigned, v);
  u = __builtin_amdgcn_permlane16(u, u, s0, s1, false, false);
  return __builtin_bit_cast(float, u);
}
__device__ __forceinline__ float rowmax16(float v) {
  v = fmaxf(v, perm16(v, 0x67452301u, 0xEFCDAB89u));  // lane ^ 1
  v = fmaxf(v, perm16(v, 0x45670123u, 0xCDEF89ABu));  // lane ^ 3
  v = fmaxf(v, perm16(v, 0x32107654u, 0xBA98FEDCu));  // lane ^ 4
  v = fmaxf(v, perm16(v, 0xFEDCBA98u, 0x76543210u));  // lane ^ 8
  return v;
}
#else
__device__ __forceinline__ float rowmax16(float v) {
#pragma unroll
  for (int off = 1; off < 16; off <<= 1) v = fmaxf(v, __shfl_xor(v, off, 32));
  return v;
}
#endif

// ---------------------------------------------------------------------------
// GEMM: Y = X @ W^T + bias.  X:[M,1024] (Tin), W:[1024,1024] f32.
// MODE 0: store f16 head-major  Yh[(b*H+h)*S + s][64]        (Q,K)
// MODE 1: store f16 transposed  Yh[(b*H+h)*64 + d][S]        (V^T)
// MODE 2: store f32 row-major   Yf[m][1024]                  (final output)
// Block = 8 waves (2M x 4N), wave tile = 32x32 (2x2 WMMA accums).
// Weights/activations are L2-resident (192 MB) so direct global->register
// fragment loads are fine; no LDS staging needed.
// ---------------------------------------------------------------------------
template <typename Tin, int MODE>
__global__ __launch_bounds__(256) void gemm_wmma(
    const Tin* __restrict__ X, const float* __restrict__ W,
    const float* __restrict__ bias, float* __restrict__ Yf,
    _Float16* __restrict__ Yh) {
  const int lane  = threadIdx.x & 31;
  const int wave  = threadIdx.x >> 5;
  const int l15   = lane & 15;
  const int hsel  = (lane >> 4) & 1;
  const int m_base = blockIdx.x * 64 + (wave >> 2) * 32;
  const int n_base = blockIdx.y * 128 + (wave & 3) * 32;

  const Tin*   xr0 = X + (size_t)(m_base + l15) * D_MODEL;
  const Tin*   xr1 = X + (size_t)(m_base + 16 + l15) * D_MODEL;
  const float* wr0 = W + (size_t)(n_base + l15) * D_MODEL;
  const float* wr1 = W + (size_t)(n_base + 16 + l15) * D_MODEL;

  v8f acc[2][2] = {};
  for (int k0 = 0; k0 < D_MODEL; k0 += 32) {
    const int ka = k0 + hsel * 8;    // A halves: [ka..ka+7], [ka+16..ka+23]
    const int kb = k0 + hsel * 8;    // B halves: [kb..kb+7], [kb+16..kb+23]
    if (k0 + 32 < D_MODEL) {         // pull next K-slab toward the WGP
      __builtin_prefetch(wr0 + k0 + 32, 0, 0);
      __builtin_prefetch(xr0 + k0 + 32, 0, 0);
    }
    v16h a0 = frag16(xr0 + ka, xr0 + ka + 16);
    v16h a1 = frag16(xr1 + ka, xr1 + ka + 16);
    v16h b0 = frag16(wr0 + kb, wr0 + kb + 16);
    v16h b1 = frag16(wr1 + kb, wr1 + kb + 16);
    acc[0][0] = wmma_f16(a0, b0, acc[0][0]);
    acc[0][1] = wmma_f16(a0, b1, acc[0][1]);
    acc[1][0] = wmma_f16(a1, b0, acc[1][0]);
    acc[1][1] = wmma_f16(a1, b1, acc[1][1]);
  }

#pragma unroll
  for (int j = 0; j < 2; ++j) {
    const int n  = n_base + j * 16 + l15;
    const float bv = bias[n];
#pragma unroll
    for (int i = 0; i < 2; ++i) {
#pragma unroll
      for (int g = 0; g < 8; ++g) {
        const int m = m_base + i * 16 + g + hsel * 8;
        const float val = acc[i][j][g] + bv;
        if (MODE == 2) {
          Yf[(size_t)m * D_MODEL + n] = val;
        } else {
          const int b = m >> 11, s = m & (SEQ - 1);
          const int h = n >> 6,  d = n & (D_HEAD - 1);
          if (MODE == 0)
            Yh[((size_t)(b * N_HEADS + h) * SEQ + s) * D_HEAD + d] = (_Float16)val;
          else
            Yh[((size_t)(b * N_HEADS + h) * D_HEAD + d) * SEQ + s] = (_Float16)val;
        }
      }
    }
  }
}

// ---------------------------------------------------------------------------
// Flash attention, causal.  Block = 4 waves, each wave owns 16 query rows.
// Q,K: [BH][S][64] f16; Vt: [BH][64][S] f16; Ctx: [B*S][1024] f16.
// Row sums are computed with an extra WMMA against an all-ones B matrix
// (rowsum lands in every lane of the matching C-fragment half-row), so the
// only cross-lane op left is the row-max permlane16 butterfly.
// ---------------------------------------------------------------------------
__global__ __launch_bounds__(128) void attn_flash(
    const _Float16* __restrict__ Q, const _Float16* __restrict__ K,
    const _Float16* __restrict__ Vt, _Float16* __restrict__ Ctx) {
  __shared__ __align__(16) _Float16 plds[4][16][72];  // P tile, padded rows

  const int lane = threadIdx.x & 31;
  const int wave = threadIdx.x >> 5;
  const int l15  = lane & 15;
  const int hsel = (lane >> 4) & 1;
  const int bh   = blockIdx.y;
  const int q0   = blockIdx.x * 64 + wave * 16;

  const _Float16* qb = Q  + (size_t)bh * SEQ * D_HEAD;
  const _Float16* kb = K  + (size_t)bh * SEQ * D_HEAD;
  const _Float16* vb = Vt + (size_t)bh * D_HEAD * SEQ;

  // Q fragments live in registers for the whole KV loop (2 x K-dim=32 chunks).
  const _Float16* qrow = qb + (size_t)(q0 + l15) * D_HEAD;
  v16h aq0 = frag16(qrow + hsel * 8,      qrow + 16 + hsel * 8);
  v16h aq1 = frag16(qrow + 32 + hsel * 8, qrow + 48 + hsel * 8);

  v16h onesf;
#pragma unroll
  for (int i = 0; i < 16; ++i) onesf[i] = (_Float16)1.0f;

  v8f o[4] = {};
  float mrow[8], lrow[8];
#pragma unroll
  for (int g = 0; g < 8; ++g) { mrow[g] = -3.0e38f; lrow[g] = 0.f; }

  for (int k0 = 0; k0 <= q0 + 15; k0 += 64) {
    // S = Q @ K^T for a 16x64 tile (4 N-subtiles of 16 keys).
    v8f s[4];
#pragma unroll
    for (int t = 0; t < 4; ++t) {
      const _Float16* krow = kb + (size_t)(k0 + t * 16 + l15) * D_HEAD;
      v16h b0 = frag16(krow + hsel * 8,      krow + 16 + hsel * 8);
      v16h b1 = frag16(krow + 32 + hsel * 8, krow + 48 + hsel * 8);
      v8f c = {};
      c = wmma_f16(aq0, b0, c);
      c = wmma_f16(aq1, b1, c);
      s[t] = c;
    }

    const bool need_mask = (k0 + 63 > q0);
#pragma unroll
    for (int t = 0; t < 4; ++t)
#pragma unroll
      for (int g = 0; g < 8; ++g) {
        float sv = s[t][g] * 0.125f;  // 1/sqrt(Dh)
        if (need_mask) {
          const int kg = k0 + t * 16 + l15;
          const int qg = q0 + g + hsel * 8;
          if (kg > qg) sv += -1e9f;   // additive causal mask, as in reference
        }
        s[t][g] = sv;
      }

    // Online softmax: row max via permlane16 butterfly, then exp.
    float mnew[8], alpha[8];
#pragma unroll
    for (int g = 0; g < 8; ++g) {
      float mt = fmaxf(fmaxf(s[0][g], s[1][g]), fmaxf(s[2][g], s[3][g]));
      mt = rowmax16(mt);
      mnew[g]  = fmaxf(mrow[g], mt);
      alpha[g] = __expf(mrow[g] - mnew[g]);
      mrow[g]  = mnew[g];
    }
#pragma unroll
    for (int t = 0; t < 4; ++t)
#pragma unroll
      for (int g = 0; g < 8; ++g)
        s[t][g] = __expf(s[t][g] - mnew[g]);
#pragma unroll
    for (int t = 0; t < 4; ++t)
#pragma unroll
      for (int g = 0; g < 8; ++g) o[t][g] *= alpha[g];

    // Shuffle P from C-layout (f32) to A-layout (f16) via wave-private LDS.
#pragma unroll
    for (int t = 0; t < 4; ++t)
#pragma unroll
      for (int g = 0; g < 8; ++g)
        plds[wave][g + hsel * 8][t * 16 + l15] = (_Float16)s[t][g];
    asm volatile("s_wait_dscnt 0" ::: "memory");
    const _Float16* prow = &plds[wave][l15][0];
    v16h ap0 = frag16(prow + hsel * 8,      prow + 16 + hsel * 8);
    v16h ap1 = frag16(prow + 32 + hsel * 8, prow + 48 + hsel * 8);

    // Row sums via WMMA against all-ones B: every lane of a half-row gets
    // rowsum(P) with the same (g,hsel) row mapping as lrow.
    v8f rs = {};
    rs = wmma_f16(ap0, onesf, rs);
    rs = wmma_f16(ap1, onesf, rs);
#pragma unroll
    for (int g = 0; g < 8; ++g) lrow[g] = lrow[g] * alpha[g] + rs[g];

    // O += P @ V (V^T rows are contiguous along the key dimension).
#pragma unroll
    for (int t = 0; t < 4; ++t) {
      const _Float16* vrow = vb + (size_t)(t * 16 + l15) * SEQ + k0;
      v16h b0 = frag16(vrow + hsel * 8,      vrow + 16 + hsel * 8);
      v16h b1 = frag16(vrow + 32 + hsel * 8, vrow + 48 + hsel * 8);
      o[t] = wmma_f16(ap0, b0, o[t]);
      o[t] = wmma_f16(ap1, b1, o[t]);
    }
  }

  // Normalize and emit context in [B*S][D_MODEL] f16 for the final GEMM.
  const int b = bh >> 4, h = bh & (N_HEADS - 1);
#pragma unroll
  for (int t = 0; t < 4; ++t)
#pragma unroll
    for (int g = 0; g < 8; ++g) {
      const int srow = q0 + g + hsel * 8;
      const int col  = h * D_HEAD + t * 16 + l15;
      const float val = o[t][g] / lrow[g];
      Ctx[(size_t)(b * SEQ + srow) * D_MODEL + col] = (_Float16)val;
    }
}

// ---------------------------------------------------------------------------
extern "C" void kernel_launch(void* const* d_in, const int* in_sizes, int n_in,
                              void* d_out, int out_size, void* d_ws, size_t ws_size,
                              hipStream_t stream) {
  const float* q  = (const float*)d_in[0];
  const float* k  = (const float*)d_in[1];
  const float* v  = (const float*)d_in[2];
  // d_in[3] = causal mask (computed analytically in-kernel)
  const float* Wq = (const float*)d_in[4];
  const float* bq = (const float*)d_in[5];
  const float* Wk = (const float*)d_in[6];
  const float* bk = (const float*)d_in[7];
  const float* Wv = (const float*)d_in[8];
  const float* bv = (const float*)d_in[9];
  const float* Wo = (const float*)d_in[10];
  const float* bo = (const float*)d_in[11];
  float* out = (float*)d_out;

  const size_t headElems = (size_t)BATCH * N_HEADS * SEQ * D_HEAD;  // 8.39M
  _Float16* qh  = (_Float16*)d_ws;
  _Float16* kh  = qh + headElems;
  _Float16* vt  = kh + headElems;
  _Float16* ctx = vt + headElems;

  dim3 gblk(256);
  dim3 ggrid(BATCH * SEQ / 64, D_MODEL / 128);
  gemm_wmma<float, 0><<<ggrid, gblk, 0, stream>>>(q, Wq, bq, nullptr, qh);
  gemm_wmma<float, 0><<<ggrid, gblk, 0, stream>>>(k, Wk, bk, nullptr, kh);
  gemm_wmma<float, 1><<<ggrid, gblk, 0, stream>>>(v, Wv, bv, nullptr, vt);

  attn_flash<<<dim3(SEQ / 64, BATCH * N_HEADS), 128, 0, stream>>>(qh, kh, vt, ctx);

  gemm_wmma<_Float16, 2><<<ggrid, gblk, 0, stream>>>(ctx, Wo, bo, out, nullptr);
}
